// Encoder_21552145891919
// MI455X (gfx1250) — compile-verified, matrix operand order checked by reading, not run on hardware
//
#include <hip/hip_runtime.h>
#include <math.h>

// ---------------------------------------------------------------------------
// Static architecture constants (match the reference)
// ---------------------------------------------------------------------------
#define PP 4
#define BB 8
#define TT 12
#define HH 160
#define WW 320
#define NHH 8
#define DU 128
#define FF 1024
#define DFF 2048
#define NTOK (BB * TT * PP)      /* 384 tokens */
#define NIMG (PP * BB * TT)      /* 384 frames */

typedef __attribute__((ext_vector_type(16))) _Float16 v16h;
typedef __attribute__((ext_vector_type(8)))  _Float16 v8h;
typedef __attribute__((ext_vector_type(8)))  float    v8f;

__device__ __forceinline__ float leaky(float x) { return x >= 0.f ? x : 0.3f * x; }

// ---------------------------------------------------------------------------
// Precision staging: activations f32 -> f16 (straight), weights f32[K,N] ->
// f16 transposed [N,K] so both GEMM operands are K-contiguous (b128 loads).
// ---------------------------------------------------------------------------
__global__ void cvt_f16_kernel(const float* __restrict__ in, _Float16* __restrict__ out, int n) {
    int i = blockIdx.x * blockDim.x + threadIdx.x;
    if (i < n) out[i] = (_Float16)in[i];
}

__global__ void cvtT_w_kernel(const float* __restrict__ W, _Float16* __restrict__ Wt,
                              int K, int N) {
    int i = blockIdx.x * blockDim.x + threadIdx.x;
    if (i >= K * N) return;
    const int k = i % K, n = i / K;
    Wt[i] = (_Float16)W[(size_t)k * N + n];
}

// xm16[b, m, t*F+f] = (f16) xe[b,t,m,f]
__global__ void transpose_xm_f16_kernel(const float* __restrict__ xe, _Float16* __restrict__ xm) {
    int idx = blockIdx.x * blockDim.x + threadIdx.x;
    const int total = BB * PP * TT * FF;
    if (idx >= total) return;
    const int f = idx % FF;
    const int t = (idx / FF) % TT;
    const int m = (idx / (FF * TT)) % PP;
    const int b = idx / (FF * TT * PP);
    xm[idx] = (_Float16)xe[(size_t)(((b * TT + t) * PP) + m) * FF + f];
}

// ---------------------------------------------------------------------------
// LDS-staged WMMA GEMM: C[M,N] = act(A16[M,K] @ Bt16[N,K]^T + bias[N])
// block = 256 threads (8 waves, 2x4 wave grid); block tile 32(M) x 128(N);
// K-step 32. LDS rows padded to 40 halves (80B) -> 16B-aligned conflict-light
// ds_load_b128 fragment reads. Each wave: 2 WMMAs/K-step, A fragment reused.
// Requires M%32==0, N%128==0, K%32==0 (all shapes here comply).
// ---------------------------------------------------------------------------
__global__ void gemm_wmma_lds_kernel(const _Float16* __restrict__ A16,
                                     const _Float16* __restrict__ Bt16,
                                     const float* __restrict__ bias,
                                     float* __restrict__ C,
                                     int M, int N, int K, int act) {
    __shared__ _Float16 Ah[32][40];
    __shared__ _Float16 Bh[128][40];
    const int tid  = threadIdx.x;
    const int lane = tid & 31;
    const int wave = tid >> 5;
    const int wm = wave >> 2, wn = wave & 3;
    const int rowB = blockIdx.y * 32;
    const int colB = blockIdx.x * 128;
    const int kb8   = (lane >> 4) << 3;            // 0 or 8
    const int mloc  = (wm << 4) + (lane & 15);
    const int nloc0 = (wn << 5) + (lane & 15);
    v8f acc0 = {}, acc1 = {};
    for (int kk = 0; kk < K; kk += 32) {
        if (tid < 128) {                            // A tile: 32x32 halves
            const int r = tid >> 2, c8 = (tid & 3) << 3;
            v8h av = *(const v8h*)(A16 + (size_t)(rowB + r) * K + kk + c8);
            *(v8h*)&Ah[r][c8] = av;
        }
#pragma unroll
        for (int j = 0; j < 2; ++j) {               // B tile: 128x32 halves
            const int idx = tid + (j << 8);
            const int nr = idx >> 2, c8 = (idx & 3) << 3;
            v8h bv = *(const v8h*)(Bt16 + (size_t)(colB + nr) * K + kk + c8);
            *(v8h*)&Bh[nr][c8] = bv;
        }
        __syncthreads();
        const v8h alo  = *(const v8h*)&Ah[mloc][kb8];
        const v8h ahi  = *(const v8h*)&Ah[mloc][kb8 + 16];
        const v8h b0lo = *(const v8h*)&Bh[nloc0][kb8];
        const v8h b0hi = *(const v8h*)&Bh[nloc0][kb8 + 16];
        const v8h b1lo = *(const v8h*)&Bh[nloc0 + 16][kb8];
        const v8h b1hi = *(const v8h*)&Bh[nloc0 + 16][kb8 + 16];
        v16h a, b0, b1;
#pragma unroll
        for (int i = 0; i < 8; ++i) {
            a[i] = alo[i];   a[i + 8] = ahi[i];
            b0[i] = b0lo[i]; b0[i + 8] = b0hi[i];
            b1[i] = b1lo[i]; b1[i + 8] = b1hi[i];
        }
        acc0 = __builtin_amdgcn_wmma_f32_16x16x32_f16(false, a, false, b0, (short)0, acc0, false, false);
        acc1 = __builtin_amdgcn_wmma_f32_16x16x32_f16(false, a, false, b1, (short)0, acc1, false, false);
        __syncthreads();
    }
    const int mrow = rowB + (wm << 4) + ((lane >> 4) << 3);
    const int n0   = colB + (wn << 5) + (lane & 15);
#pragma unroll
    for (int s = 0; s < 2; ++s) {
        const int n = n0 + s * 16;
        const float bn = bias[n];
        const v8f acc = s ? acc1 : acc0;
#pragma unroll
        for (int r = 0; r < 8; ++r) {
            float v = acc[r] + bn;
            if (act == 1) v = fmaxf(v, 0.f);
            C[(size_t)(mrow + r) * N + n] = v;
        }
    }
}

// ---------------------------------------------------------------------------
// Implicit-GEMM WMMA for the transformer value conv (3x3 SAME over T x M grid,
// 1024->1024): M'=384 tokens, N=1024, K=9*1024. Tap decode lives in the LDS
// fill: an 8-half chunk never crosses a 1024-aligned tap boundary, so it is
// one bounds check + one b128 load (zero-fill at borders). EXEC stays all-1s.
// ---------------------------------------------------------------------------
__global__ void conv_v_wmma_lds_kernel(const _Float16* __restrict__ xe16,
                                       const _Float16* __restrict__ Wt16, /* [1024, 9216] */
                                       const float* __restrict__ bias,
                                       float* __restrict__ V) {
    const int N = FF, K = 9 * FF;
    __shared__ _Float16 Ah[32][40];
    __shared__ _Float16 Bh[128][40];
    const int tid  = threadIdx.x;
    const int lane = tid & 31;
    const int wave = tid >> 5;
    const int wm = wave >> 2, wn = wave & 3;
    const int rowB = blockIdx.y * 32;
    const int colB = blockIdx.x * 128;
    const int kb8   = (lane >> 4) << 3;
    const int mloc  = (wm << 4) + (lane & 15);
    const int nloc0 = (wn << 5) + (lane & 15);
    v8f acc0 = {}, acc1 = {};
    for (int kk = 0; kk < K; kk += 32) {
        if (tid < 128) {                            // gathered A tile
            const int r = tid >> 2, c8 = (tid & 3) << 3;
            const int tok = rowB + r;
            const int bb = tok / (TT * PP), rem = tok % (TT * PP);
            const int t0 = rem >> 2, m0 = rem & 3;
            const int k = kk + c8;
            const int tap = k >> 10, cc = k & (FF - 1);
            const int t2 = t0 + tap / 3 - 1;
            const int m2 = m0 + tap % 3 - 1;
            v8h av = {};
            if ((unsigned)t2 < TT && (unsigned)m2 < PP)
                av = *(const v8h*)(xe16 + (size_t)(((bb * TT + t2) * PP) + m2) * FF + cc);
            *(v8h*)&Ah[r][c8] = av;
        }
#pragma unroll
        for (int j = 0; j < 2; ++j) {
            const int idx = tid + (j << 8);
            const int nr = idx >> 2, c8 = (idx & 3) << 3;
            v8h bv = *(const v8h*)(Wt16 + (size_t)(colB + nr) * K + kk + c8);
            *(v8h*)&Bh[nr][c8] = bv;
        }
        __syncthreads();
        const v8h alo  = *(const v8h*)&Ah[mloc][kb8];
        const v8h ahi  = *(const v8h*)&Ah[mloc][kb8 + 16];
        const v8h b0lo = *(const v8h*)&Bh[nloc0][kb8];
        const v8h b0hi = *(const v8h*)&Bh[nloc0][kb8 + 16];
        const v8h b1lo = *(const v8h*)&Bh[nloc0 + 16][kb8];
        const v8h b1hi = *(const v8h*)&Bh[nloc0 + 16][kb8 + 16];
        v16h a, b0, b1;
#pragma unroll
        for (int i = 0; i < 8; ++i) {
            a[i] = alo[i];   a[i + 8] = ahi[i];
            b0[i] = b0lo[i]; b0[i + 8] = b0hi[i];
            b1[i] = b1lo[i]; b1[i + 8] = b1hi[i];
        }
        acc0 = __builtin_amdgcn_wmma_f32_16x16x32_f16(false, a, false, b0, (short)0, acc0, false, false);
        acc1 = __builtin_amdgcn_wmma_f32_16x16x32_f16(false, a, false, b1, (short)0, acc1, false, false);
        __syncthreads();
    }
    const int mrow = rowB + (wm << 4) + ((lane >> 4) << 3);
    const int n0   = colB + (wn << 5) + (lane & 15);
#pragma unroll
    for (int s = 0; s < 2; ++s) {
        const int n = n0 + s * 16;
        const float bn = bias[n];
        const v8f acc = s ? acc1 : acc0;
#pragma unroll
        for (int r = 0; r < 8; ++r)
            V[(size_t)(mrow + r) * N + n] = acc[r] + bn;
    }
}

// ---------------------------------------------------------------------------
// Stage 0: x[B,T,H,W,P] -> xT[(p*B+b)*T+t, H, W] single channel
// ---------------------------------------------------------------------------
__global__ void transpose_x_kernel(const float* __restrict__ X, float* __restrict__ xT) {
    int idx = blockIdx.x * blockDim.x + threadIdx.x;
    const int total = NIMG * HH * WW;
    if (idx >= total) return;
    int pix = idx % (HH * WW), nimg = idx / (HH * WW);
    int h = pix / WW, w = pix % WW;
    int p = nimg / (BB * TT), b = (nimg / TT) % BB, t = nimg % TT;
    xT[idx] = X[((((size_t)(b * TT + t) * HH + h) * WW + w) * PP) + p];
}

// ---------------------------------------------------------------------------
// Fused conv3x3(SAME) + LeakyReLU(0.3) + maxpool2 + BatchNorm(inference)
// ---------------------------------------------------------------------------
__global__ void conv_pool_bn_kernel(const float* __restrict__ in,
                                    const float* __restrict__ w,
                                    const float* __restrict__ cb,
                                    const float* __restrict__ bng,
                                    const float* __restrict__ bnb,
                                    const float* __restrict__ bnm,
                                    const float* __restrict__ bnv,
                                    float* __restrict__ out,
                                    int Nimg, int H, int W, int Cin, int Cout) {
    int idx = blockIdx.x * blockDim.x + threadIdx.x;
    const int H2 = H >> 1, W2 = W >> 1;
    const int total = Nimg * H2 * W2 * Cout;
    if (idx >= total) return;
    int c = idx % Cout;
    int x2 = (idx / Cout) % W2;
    int y2 = (idx / (Cout * W2)) % H2;
    int n  = idx / (Cout * W2 * H2);
    float best = -3.0e38f;
    for (int dy = 0; dy < 2; ++dy)
    for (int dx = 0; dx < 2; ++dx) {
        const int y = y2 * 2 + dy, x = x2 * 2 + dx;
        float acc = cb[c];
        for (int ky = 0; ky < 3; ++ky) {
            const int iy = y + ky - 1;
            if (iy < 0 || iy >= H) continue;
            for (int kx = 0; kx < 3; ++kx) {
                const int ix = x + kx - 1;
                if (ix < 0 || ix >= W) continue;
                const float* ip = in + ((size_t)(n * H + iy) * W + ix) * Cin;
                const float* wp = w + (size_t)(ky * 3 + kx) * Cin * Cout + c;
                for (int ci = 0; ci < Cin; ++ci)
                    acc += ip[ci] * wp[(size_t)ci * Cout];
            }
        }
        best = fmaxf(best, leaky(acc));
    }
    out[idx] = (best - bnm[c]) * rsqrtf(bnv[c] + 1e-3f) * bng[c] + bnb[c];
}

// ---------------------------------------------------------------------------
// ConvAttention score per frame: a[r] = tanh(flat[r]@w1 + b1) @ w2 + b2
// N=16 GEMV -> AI ~ 8 flop/byte: HBM-bound, streaming VALU kernel is optimal.
// ---------------------------------------------------------------------------
__global__ void att_score_kernel(const float* __restrict__ flat,
                                 const float* __restrict__ w1,
                                 const float* __restrict__ b1,
                                 const float* __restrict__ w2,
                                 const float* __restrict__ b2,
                                 float* __restrict__ outA, int feat) {
    __shared__ float red[256];
    __shared__ float hacc[16];
    const int r = blockIdx.x;
    const float* fp = flat + (size_t)r * feat;
    float h[16];
#pragma unroll
    for (int j = 0; j < 16; ++j) h[j] = 0.f;
    for (int k = threadIdx.x; k < feat; k += 256) {
        const float v = fp[k];
        const float* wr = w1 + (size_t)k * 16;
#pragma unroll
        for (int j = 0; j < 16; ++j) h[j] += v * wr[j];
    }
    for (int j = 0; j < 16; ++j) {
        red[threadIdx.x] = h[j];
        __syncthreads();
        for (int s = 128; s > 0; s >>= 1) {
            if (threadIdx.x < s) red[threadIdx.x] += red[threadIdx.x + s];
            __syncthreads();
        }
        if (threadIdx.x == 0) hacc[j] = red[0];
        __syncthreads();
    }
    if (threadIdx.x == 0) {
        float s = b2[0];
        for (int j = 0; j < 16; ++j) s += tanhf(hacc[j] + b1[j]) * w2[j];
        outA[r] = s;
    }
}

// ---------------------------------------------------------------------------
// softmax over T + weighted sum -> skip [P,B,feat] (written into d_out slice)
// ---------------------------------------------------------------------------
__global__ void weighted_skip_kernel(const float* __restrict__ flat,
                                     const float* __restrict__ a,
                                     float* __restrict__ skip, int feat) {
    int idx = blockIdx.x * blockDim.x + threadIdx.x;
    const int total = PP * BB * feat;
    if (idx >= total) return;
    const int f = idx % feat, pb = idx / feat;
    const float* ap = a + pb * TT;
    float mx = -3.0e38f;
    for (int t = 0; t < TT; ++t) mx = fmaxf(mx, ap[t]);
    float e[TT], den = 0.f;
    for (int t = 0; t < TT; ++t) { e[t] = expf(ap[t] - mx); den += e[t]; }
    float s = 0.f;
    const float* fp = flat + (size_t)pb * TT * feat + f;
    for (int t = 0; t < TT; ++t) s += e[t] * fp[(size_t)t * feat];
    skip[idx] = s / den;
}

// ---------------------------------------------------------------------------
// convf 5x5/stride5/VALID + leaky -> xe[B,T,P,F] with positional encoding
// ---------------------------------------------------------------------------
__global__ void convf_pe_kernel(const float* __restrict__ y3,  /* [384,20,40,16] */
                                const float* __restrict__ w,   /* [5,5,16,32] */
                                const float* __restrict__ cb,
                                float* __restrict__ xe) {      /* [B,T,P,F] */
    int idx = blockIdx.x * blockDim.x + threadIdx.x;
    const int total = NIMG * 4 * 8 * 32;
    if (idx >= total) return;
    const int c  = idx % 32;
    const int xx = (idx / 32) % 8;
    const int yy = (idx / 256) % 4;
    const int n  = idx / 1024;
    float acc = cb[c];
    const int y0 = yy * 5, x0 = xx * 5;
    for (int ky = 0; ky < 5; ++ky)
    for (int kx = 0; kx < 5; ++kx) {
        const float* ip = y3 + ((size_t)(n * 20 + y0 + ky) * 40 + x0 + kx) * 16;
        const float* wp = w + (size_t)(ky * 5 + kx) * 16 * 32 + c;
        for (int ci = 0; ci < 16; ++ci) acc += ip[ci] * wp[(size_t)ci * 32];
    }
    acc = leaky(acc);
    const int f = (yy * 8 + xx) * 32 + c;
    const int p = n / (BB * TT), b = (n / TT) % BB, t = n % TT;
    const float ang = (float)t / __powf(10000.f, (2.f * floorf(f * 0.5f)) / (float)FF);
    const float pe  = (f & 1) ? __cosf(ang) : __sinf(ang);
    xe[(size_t)(((b * TT + t) * PP) + p) * FF + f] = acc + pe;
}

// logits[b,h,q,k] = Q[b,q,h,:] . K[b,k,h,:] * scale   (Q,K: [B,S,NH,U])
__global__ void attn_logits_kernel(const float* __restrict__ Q,
                                   const float* __restrict__ Kt,
                                   float* __restrict__ logits,
                                   int S, float scale) {
    int idx = blockIdx.x * blockDim.x + threadIdx.x;
    const int total = BB * NHH * S * S;
    if (idx >= total) return;
    const int kk = idx % S;
    const int q  = (idx / S) % S;
    const int h  = (idx / (S * S)) % NHH;
    const int b  = idx / (S * S * NHH);
    const float* qp = Q  + (size_t)((b * S + q) * NHH + h) * DU;
    const float* kp = Kt + (size_t)((b * S + kk) * NHH + h) * DU;
    float s = 0.f;
    for (int u = 0; u < DU; ++u) s += qp[u] * kp[u];
    logits[idx] = s * scale;
}

__global__ void softmax_rows_kernel(float* __restrict__ d, int rows, int S) {
    int r = blockIdx.x * blockDim.x + threadIdx.x;
    if (r >= rows) return;
    float* p = d + (size_t)r * S;
    float mx = -3.0e38f;
    for (int i = 0; i < S; ++i) mx = fmaxf(mx, p[i]);
    float den = 0.f;
    for (int i = 0; i < S; ++i) { float e = expf(p[i] - mx); p[i] = e; den += e; }
    const float inv = 1.f / den;
    for (int i = 0; i < S; ++i) p[i] *= inv;
}

// tmp[b,h,q,n,d] = sum_k at[b,h,q,k] * V[b,k,n,h,d]
__global__ void attn_tv_kernel(const float* __restrict__ at,
                               const float* __restrict__ V,
                               float* __restrict__ tmp) {
    int idx = blockIdx.x * blockDim.x + threadIdx.x;
    const int total = BB * NHH * TT * PP * DU;
    if (idx >= total) return;
    const int d = idx % DU;
    const int n = (idx / DU) % PP;
    const int q = (idx / (DU * PP)) % TT;
    const int h = (idx / (DU * PP * TT)) % NHH;
    const int b = idx / (DU * PP * TT * NHH);
    const float* ap = at + (size_t)((b * NHH + h) * TT + q) * TT;
    float s = 0.f;
    for (int k = 0; k < TT; ++k)
        s += ap[k] * V[(size_t)(((b * TT + k) * PP + n) * NHH + h) * DU + d];
    tmp[idx] = s;
}

// o[b,q,m, h*DU+d] = sum_n am[b,h,m,n] * tmp[b,h,q,n,d]
__global__ void attn_o_kernel(const float* __restrict__ am,
                              const float* __restrict__ tmp,
                              float* __restrict__ o) {
    int idx = blockIdx.x * blockDim.x + threadIdx.x;
    const int total = BB * TT * PP * NHH * DU;
    if (idx >= total) return;
    const int d = idx % DU;
    const int h = (idx / DU) % NHH;
    const int m = (idx / (DU * NHH)) % PP;
    const int q = (idx / (DU * NHH * PP)) % TT;
    const int b = idx / (DU * NHH * PP * TT);
    float s = 0.f;
    for (int n = 0; n < PP; ++n)
        s += am[(size_t)((b * NHH + h) * PP + m) * PP + n] *
             tmp[(size_t)((((b * NHH + h) * TT + q) * PP) + n) * DU + d];
    o[(size_t)(((b * TT + q) * PP) + m) * FF + h * DU + d] = s;
}

// x = LayerNorm(x + r) * g + b   (one 256-thread block per 1024-wide token)
__global__ void add_ln_kernel(float* __restrict__ x, const float* __restrict__ r,
                              const float* __restrict__ g, const float* __restrict__ b) {
    __shared__ float red[256];
    const int tok = blockIdx.x;
    float* xp = x + (size_t)tok * FF;
    const float* rp = r + (size_t)tok * FF;
    float v[4];
    float s = 0.f;
#pragma unroll
    for (int i = 0; i < 4; ++i) { v[i] = xp[threadIdx.x + 256 * i] + rp[threadIdx.x + 256 * i]; s += v[i]; }
    red[threadIdx.x] = s; __syncthreads();
    for (int st = 128; st > 0; st >>= 1) {
        if (threadIdx.x < st) red[threadIdx.x] += red[threadIdx.x + st];
        __syncthreads();
    }
    const float mean = red[0] * (1.f / FF);
    __syncthreads();
    s = 0.f;
#pragma unroll
    for (int i = 0; i < 4; ++i) { const float d = v[i] - mean; s += d * d; }
    red[threadIdx.x] = s; __syncthreads();
    for (int st = 128; st > 0; st >>= 1) {
        if (threadIdx.x < st) red[threadIdx.x] += red[threadIdx.x + st];
        __syncthreads();
    }
    const float inv = rsqrtf(red[0] * (1.f / FF) + 1e-6f);
#pragma unroll
    for (int i = 0; i < 4; ++i) {
        const int c = threadIdx.x + 256 * i;
        xp[c] = (v[i] - mean) * inv * g[c] + b[c];
    }
}

__global__ void copy_kernel(const float* __restrict__ src, float* __restrict__ dst, int n) {
    int i = blockIdx.x * blockDim.x + threadIdx.x;
    if (i < n) dst[i] = src[i];
}

// ---------------------------------------------------------------------------
// Host orchestration
// ---------------------------------------------------------------------------
struct Dense { const float *b, *w; };
struct AttP  { const float *b1, *b2, *w1, *w2; };
struct BnP   { const float *bt, *g, *m, *v; };
struct LayerP {
    Dense ff1, ff2, km, kt;
    const float *ln1b, *ln1g, *ln2b, *ln2g;
    Dense out, qm, qt, v;
};

extern "C" void kernel_launch(void* const* d_in, const int* in_sizes, int n_in,
                              void* d_out, int out_size, void* d_ws, size_t ws_size,
                              hipStream_t stream) {
    (void)in_sizes; (void)n_in; (void)out_size; (void)ws_size;
    // -------- input leaves (jax pytree order: dict keys sorted within params) --
    const float* X = (const float*)d_in[0];
    int ix = 1;
    auto NXT = [&]() { return (const float*)d_in[ix++]; };
    AttP att1, att2, att3;
    BnP  bn1, bn2, bn3;
    Dense conv1, conv2, conv3, convf;
    LayerP lay[2];
    AttP* atts[3] = {&att1, &att2, &att3};
    for (int i = 0; i < 3; ++i) { atts[i]->b1 = NXT(); atts[i]->b2 = NXT(); atts[i]->w1 = NXT(); atts[i]->w2 = NXT(); }
    BnP* bns[3] = {&bn1, &bn2, &bn3};
    for (int i = 0; i < 3; ++i) { bns[i]->bt = NXT(); bns[i]->g = NXT(); bns[i]->m = NXT(); bns[i]->v = NXT(); }
    Dense* cvs[4] = {&conv1, &conv2, &conv3, &convf};
    for (int i = 0; i < 4; ++i) { cvs[i]->b = NXT(); cvs[i]->w = NXT(); }
    for (int l = 0; l < 2; ++l) {
        lay[l].ff1.b = NXT(); lay[l].ff1.w = NXT();
        lay[l].ff2.b = NXT(); lay[l].ff2.w = NXT();
        lay[l].km.b  = NXT(); lay[l].km.w  = NXT();
        lay[l].kt.b  = NXT(); lay[l].kt.w  = NXT();
        lay[l].ln1b  = NXT(); lay[l].ln1g  = NXT();
        lay[l].ln2b  = NXT(); lay[l].ln2g  = NXT();
        lay[l].out.b = NXT(); lay[l].out.w = NXT();
        lay[l].qm.b  = NXT(); lay[l].qm.w  = NXT();
        lay[l].qt.b  = NXT(); lay[l].qt.w  = NXT();
        lay[l].v.b   = NXT(); lay[l].v.w   = NXT();
    }

    // -------- workspace: two overlapped 19.66M-float regions -------------------
    float* ws = (float*)d_ws;
    const size_t R = 19660800;            // region size (= |xT| = |y1|)
    float* reg0 = ws;                     // xT during stem; reused after
    float* y1   = ws + R;                 // region1: y1 during stem; f16 scratch after
    float* xT   = reg0;
    size_t off = 0;
    float* y2    = reg0 + off; off += 9830400;
    float* y3    = reg0 + off; off += 4915200;
    float* xe    = reg0 + off; off += 393216;
    float* qtb   = reg0 + off; off += 98304;
    float* ktb   = reg0 + off; off += 98304;
    float* qmb   = reg0 + off; off += 32768;
    float* kmb   = reg0 + off; off += 32768;
    float* vbuf  = reg0 + off; off += 393216;
    float* atb   = reg0 + off; off += 9216;
    float* amb   = reg0 + off; off += 1024;
    float* tmp   = reg0 + off; off += 393216;
    float* obuf  = reg0 + off; off += 393216;
    float* obuf2 = reg0 + off; off += 393216;
    float* ffh   = reg0 + off; off += 786432;
    float* ascore = reg0 + off; off += 512;

    // f16 scratch lives in region1 (y1 is dead once y2 exists; all GEMMs are
    // post-stem). Sizes in halves; worst-case total ~14.5M halves << region1.
    _Float16* f16b = (_Float16*)y1;
    _Float16* wt16 = f16b;                         // up to 12,582,912 halves
    _Float16* xe16 = f16b + 12582912;              // 393,216
    _Float16* xm16 = xe16 + 393216;                // 393,216
    _Float16* ob16 = xm16 + 393216;                // 393,216
    _Float16* fh16 = ob16 + 393216;                // 786,432

    float* Out = (float*)d_out;
    float* outXE    = Out;
    float* outSkip1 = Out + 393216;
    float* outSkip2 = outSkip1 + 1638400;
    float* outSkip3 = outSkip2 + 819200;

    auto g1 = [](int n) { return dim3((unsigned)((n + 255) / 256)); };
    const dim3 b256(256);
    const float scale = 0.088388347648318440550f;   // 1/sqrt(128)

    // ---------------- conv stem ----------------
    transpose_x_kernel<<<g1(NIMG * HH * WW), b256, 0, stream>>>(X, xT);

    conv_pool_bn_kernel<<<g1(NIMG * 80 * 160 * 4), b256, 0, stream>>>(
        xT, conv1.w, conv1.b, bn1.g, bn1.bt, bn1.m, bn1.v, y1, NIMG, HH, WW, 1, 4);
    att_score_kernel<<<dim3(NIMG), b256, 0, stream>>>(y1, att1.w1, att1.b1, att1.w2, att1.b2, ascore, 80 * 160 * 4);
    weighted_skip_kernel<<<g1(PP * BB * 80 * 160 * 4), b256, 0, stream>>>(y1, ascore, outSkip1, 80 * 160 * 4);

    conv_pool_bn_kernel<<<g1(NIMG * 40 * 80 * 8), b256, 0, stream>>>(
        y1, conv2.w, conv2.b, bn2.g, bn2.bt, bn2.m, bn2.v, y2, NIMG, 80, 160, 4, 8);
    att_score_kernel<<<dim3(NIMG), b256, 0, stream>>>(y2, att2.w1, att2.b1, att2.w2, att2.b2, ascore, 40 * 80 * 8);
    weighted_skip_kernel<<<g1(PP * BB * 40 * 80 * 8), b256, 0, stream>>>(y2, ascore, outSkip2, 40 * 80 * 8);

    conv_pool_bn_kernel<<<g1(NIMG * 20 * 40 * 16), b256, 0, stream>>>(
        y2, conv3.w, conv3.b, bn3.g, bn3.bt, bn3.m, bn3.v, y3, NIMG, 40, 80, 8, 16);
    att_score_kernel<<<dim3(NIMG), b256, 0, stream>>>(y3, att3.w1, att3.b1, att3.w2, att3.b2, ascore, 20 * 40 * 16);
    weighted_skip_kernel<<<g1(PP * BB * 20 * 40 * 16), b256, 0, stream>>>(y3, ascore, outSkip3, 20 * 40 * 16);

    convf_pe_kernel<<<g1(NIMG * 1024), b256, 0, stream>>>(y3, convf.w, convf.b, xe);

    // ---------------- transformer layers (LDS-staged WMMA GEMMs) ----------------
    const dim3 bGem(256);
    for (int l = 0; l < 2; ++l) {
        const LayerP& L = lay[l];
        // stage activations to f16 once per consumer group
        cvt_f16_kernel<<<g1(NTOK * FF), b256, 0, stream>>>(xe, xe16, NTOK * FF);
        transpose_xm_f16_kernel<<<g1(NTOK * FF), b256, 0, stream>>>(xe, xm16);
        // time-axis q/k: [96,4096] @ [4096,1024]
        cvtT_w_kernel<<<g1(4096 * 1024), b256, 0, stream>>>(L.qt.w, wt16, 4096, 1024);
        gemm_wmma_lds_kernel<<<dim3(8, 3), bGem, 0, stream>>>(xe16, wt16, L.qt.b, qtb, 96, 1024, 4096, 0);
        cvtT_w_kernel<<<g1(4096 * 1024), b256, 0, stream>>>(L.kt.w, wt16, 4096, 1024);
        gemm_wmma_lds_kernel<<<dim3(8, 3), bGem, 0, stream>>>(xe16, wt16, L.kt.b, ktb, 96, 1024, 4096, 0);
        // measure-axis q/k: [32,12288] @ [12288,1024]
        cvtT_w_kernel<<<g1(12288 * 1024), b256, 0, stream>>>(L.qm.w, wt16, 12288, 1024);
        gemm_wmma_lds_kernel<<<dim3(8, 1), bGem, 0, stream>>>(xm16, wt16, L.qm.b, qmb, 32, 1024, 12288, 0);
        cvtT_w_kernel<<<g1(12288 * 1024), b256, 0, stream>>>(L.km.w, wt16, 12288, 1024);
        gemm_wmma_lds_kernel<<<dim3(8, 1), bGem, 0, stream>>>(xm16, wt16, L.km.b, kmb, 32, 1024, 12288, 0);
        // value conv as implicit GEMM: 384 x 1024 x 9216
        cvtT_w_kernel<<<g1(9216 * 1024), b256, 0, stream>>>(L.v.w, wt16, 9216, 1024);
        conv_v_wmma_lds_kernel<<<dim3(8, 12), bGem, 0, stream>>>(xe16, wt16, L.v.b, vbuf);
        // decomposed attentions (tiny: VALU)
        attn_logits_kernel<<<g1(BB * NHH * TT * TT), b256, 0, stream>>>(qtb, ktb, atb, TT, scale);
        softmax_rows_kernel<<<g1(BB * NHH * TT), b256, 0, stream>>>(atb, BB * NHH * TT, TT);
        attn_logits_kernel<<<g1(BB * NHH * PP * PP), b256, 0, stream>>>(qmb, kmb, amb, PP, scale);
        softmax_rows_kernel<<<g1(BB * NHH * PP), b256, 0, stream>>>(amb, BB * NHH * PP, PP);
        attn_tv_kernel<<<g1(BB * NHH * TT * PP * DU), b256, 0, stream>>>(atb, vbuf, tmp);
        attn_o_kernel<<<g1(BB * TT * PP * NHH * DU), b256, 0, stream>>>(amb, tmp, obuf);
        // output projection + residual LN1
        cvt_f16_kernel<<<g1(NTOK * FF), b256, 0, stream>>>(obuf, ob16, NTOK * FF);
        cvtT_w_kernel<<<g1(1024 * 1024), b256, 0, stream>>>(L.out.w, wt16, 1024, 1024);
        gemm_wmma_lds_kernel<<<dim3(8, 12), bGem, 0, stream>>>(ob16, wt16, L.out.b, obuf2, NTOK, 1024, 1024, 0);
        add_ln_kernel<<<dim3(NTOK), b256, 0, stream>>>(xe, obuf2, L.ln1g, L.ln1b);
        // FFN + residual LN2
        cvt_f16_kernel<<<g1(NTOK * FF), b256, 0, stream>>>(xe, xe16, NTOK * FF);
        cvtT_w_kernel<<<g1(1024 * 2048), b256, 0, stream>>>(L.ff1.w, wt16, 1024, 2048);
        gemm_wmma_lds_kernel<<<dim3(16, 12), bGem, 0, stream>>>(xe16, wt16, L.ff1.b, ffh, NTOK, 2048, 1024, 1);
        cvt_f16_kernel<<<g1(NTOK * DFF), b256, 0, stream>>>(ffh, fh16, NTOK * DFF);
        cvtT_w_kernel<<<g1(2048 * 1024), b256, 0, stream>>>(L.ff2.w, wt16, 2048, 1024);
        gemm_wmma_lds_kernel<<<dim3(8, 12), bGem, 0, stream>>>(fh16, wt16, L.ff2.b, obuf2, NTOK, 1024, 2048, 0);
        add_ln_kernel<<<dim3(NTOK), b256, 0, stream>>>(xe, obuf2, L.ln2g, L.ln2b);
    }

    copy_kernel<<<g1(393216), b256, 0, stream>>>(xe, outXE, 393216);
}
